// NearestEmbed_60464549593490
// MI455X (gfx1250) — compile-verified, matrix operand order checked by reading, not run on hardware
//
#include <hip/hip_runtime.h>
#include <hip/hip_bf16.h>

// ---------------------------------------------------------------------------
// VQ nearest-embedding for MI455X (gfx1250, wave32, WMMA + async LDS DMA).
//   x:   (B=64, D=256, H=32, W=32) f32
//   emb: (D=256, K=1024) f32
// out0: quantized (B, D, H, W) f32   (16,777,216 floats)
// out1: argmin    (B, H, W)          (65,536 values, written as float)
//
// Score GEMM runs on v_wmma_f32_16x16x32_f16 (f16 in, f32 accumulate).
// B fragments are staged per-block into double-buffered LDS with
// global_load_async_to_lds_b128 (ASYNCcnt) so the DMA of chunk c+1 overlaps
// the WMMA burst on chunk c, and 16 waves share one copy of each B chunk
// (16x less L2 traffic than per-wave streaming).
// Quantized output is gathered from the ORIGINAL f32 codebook (exact).
// ---------------------------------------------------------------------------

typedef __attribute__((ext_vector_type(16))) _Float16 v16h;
typedef __attribute__((ext_vector_type(8)))  float    v8f;

#define BB   64
#define DD   256
#define HH   32
#define WW   32
#define HW   1024           // H*W
#define KK   1024
#define NN   65536          // B*H*W

#define CHUNK_HALVES 4096   // one k-chunk of B: 8 frags * 32 lanes * 16 halves
#define CHUNK_BYTES  8192

// CDNA5 16-bit A-matrix (16x32) per-lane element map (ISA 7.12.2):
//   lanes 0-15 : row M = lane,    halves h -> K = (h<8 ? h : h+8)
//   lanes 16-31: row M = lane-16, halves h -> K = (h<8 ? h : h+8) + 8
// Dense 32x16 B assumed to mirror this (col N = lane%16, same K map);
// pack_emb materializes emb in that exact fragment order.
__device__ __forceinline__ int klocal(int h, int lane) {
    return (h & 7) + ((h >> 3) << 4) + ((lane >> 4) << 3);
}

// ---------------------------------------------------------------------------
// Kernel 1: pack emb (f32, row-major D x K) into f16 WMMA-B fragment order.
// embW index = (((c*8 + f)*32 + lane)*16 + h)
// ---------------------------------------------------------------------------
__global__ __launch_bounds__(256) void pack_emb(const float* __restrict__ emb,
                                                _Float16* __restrict__ embW) {
    int tid = blockIdx.x * 256 + threadIdx.x;      // 0 .. D*K-1
    int h    = tid & 15;
    int lane = (tid >> 4) & 31;
    int f    = (tid >> 9) & 7;
    int c    = tid >> 12;
    int d    = 32 * f + klocal(h, lane);
    int kcol = 16 * c + (lane & 15);
    embW[tid] = (_Float16)emb[d * KK + kcol];
}

// ---------------------------------------------------------------------------
// Kernel 2: enormH[k] = 0.5 * sum_d emb[d,k]^2   (f32)
// ---------------------------------------------------------------------------
__global__ __launch_bounds__(256) void enorm_kernel(const float* __restrict__ emb,
                                                    float* __restrict__ enormH) {
    int k = blockIdx.x * 256 + threadIdx.x;        // 0 .. K-1
    float s = 0.0f;
    for (int d = 0; d < DD; ++d) {
        float e = emb[d * KK + k];
        s += e * e;
    }
    enormH[k] = 0.5f * s;
}

// Async DMA of one 16-byte segment: memory -> LDS, tracked by ASYNCcnt.
// INST_OFFSET applies to both LDS and global addresses (ISA 10, async ops),
// but we fold offsets into the operands so one template suffices.
__device__ __forceinline__ void async_b128_to_lds(const void* gsrc,
                                                  unsigned goff,
                                                  unsigned ldsAddr) {
    asm volatile("global_load_async_to_lds_b128 %0, %1, %2"
                 :: "v"(ldsAddr), "v"(goff), "s"(gsrc)
                 : "memory");
}

__device__ __forceinline__ void wait_asynccnt0() {
    asm volatile("s_wait_asynccnt 0x0" ::: "memory");
}

// ---------------------------------------------------------------------------
// Kernel 3: main. One wave handles 16 consecutive latents; 16 waves per block
// share double-buffered B chunks in LDS. HW=1024 is a multiple of 16, so a
// 16-row tile never crosses a batch boundary.
// ---------------------------------------------------------------------------
__global__ __launch_bounds__(512) void vq_main(const float*    __restrict__ x,
                                               const float*    __restrict__ emb,
                                               const _Float16* __restrict__ embW,
                                               const float*    __restrict__ enormH,
                                               float* __restrict__ outQ,
                                               float* __restrict__ outIdx) {
    __shared__ __align__(32) _Float16 sB[2][CHUNK_HALVES];   // 2 x 8KB
    __shared__ int winners[16][16];

    const int t    = threadIdx.x;                  // 0..511
    const int lane = t & 31;
    const int wave = t >> 5;                       // 0..15
    const int tile = blockIdx.x * 16 + wave;       // 16-row tile index
    const int n0   = tile * 16;
    const int b    = n0 >> 10;                     // n0 / HW
    const int hw0  = n0 & 1023;                    // n0 % HW
    const int m    = lane & 15;                    // row this lane feeds

    // --- Kick off async DMA of B chunk 0 while we load A fragments --------
    // Each of the 512 threads moves one 16B segment (512*16B = 8KB chunk).
    const unsigned ldsBase0 = (unsigned)(size_t)(&sB[0][0]);
    const unsigned ldsBase1 = (unsigned)(size_t)(&sB[1][0]);
    async_b128_to_lds(embW, (unsigned)(t * 16), ldsBase0 + (unsigned)(t * 16));

    // --- Load A fragments once: 16 rows x full D=256 -> 8 x v16h ----------
    const float* xrow = x + (size_t)b * DD * HW + hw0 + m; // element (row, d=0)
    v16h af[8];
#pragma unroll
    for (int f = 0; f < 8; ++f) {
#pragma unroll
        for (int h = 0; h < 16; ++h) {
            int d = 32 * f + klocal(h, lane);
            af[f][h] = (_Float16)xrow[(size_t)d * HW];
        }
    }

    float bv[8];
    int   bi[8];
#pragma unroll
    for (int r = 0; r < 8; ++r) { bv[r] = -3.4e38f; bi[r] = 0; }

    wait_asynccnt0();
    __syncthreads();                               // chunk 0 resident in sB[0]

    // --- Running argmax over K in 16-column chunks ------------------------
    for (int c = 0; c < KK / 16; ++c) {
        const int p = c & 1;

        // Start DMA of chunk c+1 into the other buffer; overlaps the WMMAs.
        if (c + 1 < KK / 16) {
            unsigned dst = (p ? ldsBase0 : ldsBase1) + (unsigned)(t * 16);
            async_b128_to_lds(embW + (size_t)(c + 1) * CHUNK_HALVES,
                              (unsigned)(t * 16), dst);
        }

        const int kcol = 16 * c + m;               // this lane's column
        const float en = enormH[kcol];

        v8f acc = {};
        const _Float16* bbase = &sB[p][(size_t)lane * 16];
#pragma unroll
        for (int f = 0; f < 8; ++f) {
            v16h bf = *(const v16h*)(bbase + (size_t)f * 512);  // ds_load x2
            acc = __builtin_amdgcn_wmma_f32_16x16x32_f16(
                false, af[f], false, bf, (short)0, acc, false, false);
        }
        // C layout: lane owns col (lane&15), rows r + 8*(lane>>4), r=0..7
#pragma unroll
        for (int r = 0; r < 8; ++r) {
            float s = acc[r] - en;                 // x.e - 0.5||e||^2
            if (s > bv[r]) { bv[r] = s; bi[r] = kcol; }
        }

        wait_asynccnt0();                          // next buffer fully written
        __syncthreads();                           // all waves done with sB[p]
    }

    // --- Argmax reduction across the 16 lanes holding each row ------------
#pragma unroll
    for (int off = 1; off < 16; off <<= 1) {
#pragma unroll
        for (int r = 0; r < 8; ++r) {
            float ov = __shfl_xor(bv[r], off, 32);
            int   oi = __shfl_xor(bi[r], off, 32);
            if (ov > bv[r] || (ov == bv[r] && oi < bi[r])) { bv[r] = ov; bi[r] = oi; }
        }
    }

    // --- Exchange winners via LDS so every lane knows row (lane&15) -------
    if (lane == 0) {
#pragma unroll
        for (int r = 0; r < 8; ++r) winners[wave][r] = bi[r];        // rows 0-7
    }
    if (lane == 16) {
#pragma unroll
        for (int r = 0; r < 8; ++r) winners[wave][8 + r] = bi[r];    // rows 8-15
    }
    __syncthreads();
    const int kmine = winners[wave][m];

    // --- Write argmin plane (as float; k < 1024 is exactly representable) -
    if (lane < 16) outIdx[n0 + lane] = (float)kmine;

    // --- Write quantized vectors from the ORIGINAL f32 codebook -----------
    // For a fixed d, lanes 0-15 write 16 consecutive floats (hw0..hw0+15):
    // fully coalesced 64B stores; lanes 16-31 handle the next d-plane.
    float* outBase = outQ + (size_t)b * DD * HW + hw0 + m;
#pragma unroll 4
    for (int dd = 0; dd < DD / 2; ++dd) {
        int d = 2 * dd + (lane >> 4);
        outBase[(size_t)d * HW] = emb[(size_t)d * KK + kmine];
    }
}

// ---------------------------------------------------------------------------
extern "C" void kernel_launch(void* const* d_in, const int* in_sizes, int n_in,
                              void* d_out, int out_size, void* d_ws, size_t ws_size,
                              hipStream_t stream) {
    (void)in_sizes; (void)n_in; (void)out_size; (void)ws_size;
    const float* x   = (const float*)d_in[0];   // (B, D, H, W) f32
    const float* emb = (const float*)d_in[1];   // (D, K) f32

    _Float16* embW   = (_Float16*)d_ws;                                   // 512 KB
    float*    enormH = (float*)((char*)d_ws + (size_t)DD * KK * sizeof(_Float16));

    float* outQ   = (float*)d_out;
    float* outIdx = outQ + (size_t)BB * DD * HH * WW;                     // second output

    pack_emb<<<(DD * KK) / 256, 256, 0, stream>>>(emb, embW);
    enorm_kernel<<<KK / 256, 256, 0, stream>>>(emb, enormH);
    vq_main<<<NN / 256, 512, 0, stream>>>(x, emb, embW, enormH, outQ, outIdx);
}
